// VectorQuantizer_14723147891346
// MI455X (gfx1250) — matrix operand order repaired and checked by reading, not compiled, not run on hardware
//
#include <hip/hip_runtime.h>
#include <hip/hip_bf16.h>

typedef __attribute__((ext_vector_type(16))) _Float16 v16h;
typedef __attribute__((ext_vector_type(8)))  _Float16 v8h;
typedef __attribute__((ext_vector_type(8)))  float    v8f;

#define B_    32
#define C_    256
#define H_    32
#define W_    32
#define HW_   (H_ * W_)          // 1024
#define NPIX  (B_ * HW_)         // 32768
#define KCB   1024
#define NELEM (B_ * C_ * HW_)    // 8388608

// d_out layout (floats): [u | z_train | vq_loss | indices]
#define OUT_U    0
#define OUT_ZT   (NELEM)
#define OUT_LOSS (2 * NELEM)
#define OUT_IDX  (2 * NELEM + 1)

// workspace layout (bytes)
#define WS_U16    0                               // NPIX*C_ halfs = 16 MB
#define WS_CB16   (WS_U16 + NPIX * C_ * 2)        // 512 KB
#define WS_CBNORM (WS_CB16 + KCB * C_ * 2)        // 4 KB
#define WS_IDX    (WS_CBNORM + KCB * 4)           // 128 KB
#define WS_LOSS   (WS_IDX + NPIX * 4)             // 4 B

// sched_group_barrier masks
#define SGB_WMMA 0x008
#define SGB_VMEM_READ 0x020

// ---------------------------------------------------------------------------
// 1) codebook -> f16 copy + fp32 row norms; also zero the loss accumulator
// ---------------------------------------------------------------------------
__global__ void vq_prep_cb(const float* __restrict__ cb,
                           _Float16* __restrict__ cb16,
                           float* __restrict__ cbnorm,
                           float* __restrict__ loss_acc) {
    __shared__ float red[256];
    int row = blockIdx.x;
    int t   = threadIdx.x;
    float v = cb[row * C_ + t];
    cb16[row * C_ + t] = (_Float16)v;
    red[t] = v * v;
    __syncthreads();
    for (int s = 128; s > 0; s >>= 1) {
        if (t < s) red[t] += red[t + s];
        __syncthreads();
    }
    if (t == 0) cbnorm[row] = red[0];
    if (row == 0 && t == 0) *loss_acc = 0.0f;
}

// ---------------------------------------------------------------------------
// 2) u [B][C][HW] fp32  ->  flat16 [NPIX][C] f16 (LDS 32x32 tile transpose)
// ---------------------------------------------------------------------------
__global__ void vq_transpose_u(const float* __restrict__ u,
                               _Float16* __restrict__ u16) {
    __shared__ float tile[32][33];
    int p0 = blockIdx.x * 32;       // pixel (hw) tile
    int c0 = blockIdx.y * 32;       // channel tile
    int b  = blockIdx.z;
    int tx = threadIdx.x;           // 0..31
    int ty = threadIdx.y;           // 0..7
    const float* ub = u + (size_t)b * C_ * HW_;
#pragma unroll
    for (int i = 0; i < 4; ++i) {
        int c = c0 + ty + i * 8;
        tile[ty + i * 8][tx] = ub[(size_t)c * HW_ + p0 + tx];   // coalesced in hw
    }
    __syncthreads();
#pragma unroll
    for (int i = 0; i < 4; ++i) {
        int p = p0 + ty + i * 8;
        // row = global pixel, col = channel; coalesced in channel
        u16[((size_t)(b * HW_ + p)) * C_ + c0 + tx] = (_Float16)tile[tx][ty + i * 8];
    }
}

// ---------------------------------------------------------------------------
// 3) WMMA score GEMM + fused argmin.
//    Each wave: 32 pixels x all 1024 codes, K = 256 channels.
//    Two A panels (2 x 16x256 f16) resident in VGPRs share every B fragment
//    (1 WMMA per b128 B load). sched_group_barrier pins the issue order to
//    [5 VMEM] ([2 WMMA][2 VMEM])x6 [2 WMMA][2 WMMA], forcing the pre-RA
//    scheduler to keep two B fragments in flight so WMMA pairs overlap the
//    next fragment's loads instead of waiting loadcnt==0 before every pair.
// ---------------------------------------------------------------------------
__global__ void __launch_bounds__(256)
vq_main(const _Float16* __restrict__ u16,
        const _Float16* __restrict__ cb16,
        const float* __restrict__ cbnorm,
        int* __restrict__ idx_ws,
        float* __restrict__ out_idx) {
    int wave = threadIdx.x >> 5;
    int lane = threadIdx.x & 31;
    int hi   = lane >> 4;          // 0: lanes 0-15, 1: lanes 16-31
    int l15  = lane & 15;
    int tile = blockIdx.x * 8 + wave;          // 32-pixel tile id

    // ---- load A panels: 2 x (16 pixels x 256 channels), ISA 16-bit A layout
    // lanes 0-15 (M=l15): VGPR0-3 = K k0..k0+7, VGPR4-7 = K k0+16..k0+23
    // lanes 16-31        : VGPR0-3 = K k0+8..k0+15, VGPR4-7 = K k0+24..k0+31
    v16h A0[8], A1[8];
    const _Float16* arow0 = u16 + ((size_t)(tile * 32 + l15)) * C_;
    const _Float16* arow1 = arow0 + 16 * C_;
#pragma unroll
    for (int kk = 0; kk < 8; ++kk) {
        int k0 = kk * 32;
        v8h lo0 = *(const v8h*)(arow0 + k0 + hi * 8);
        v8h hi0 = *(const v8h*)(arow0 + k0 + 16 + hi * 8);
        v8h lo1 = *(const v8h*)(arow1 + k0 + hi * 8);
        v8h hi1 = *(const v8h*)(arow1 + k0 + 16 + hi * 8);
#pragma unroll
        for (int j = 0; j < 8; ++j) {
            A0[kk][j] = lo0[j]; A0[kk][j + 8] = hi0[j];
            A1[kk][j] = lo1[j]; A1[kk][j + 8] = hi1[j];
        }
    }

    float best[16];
    int   bidx[16];
#pragma unroll
    for (int r = 0; r < 16; ++r) { best[r] = 3.4e38f; bidx[r] = 0; }

    // ---- B tile: 32(K) x 16 codes; ISA 16-bit B layout:
// lanes 0-15 (N=l15) hold K k0..k0+7, k0+16..k0+23, lanes 16-31 hold K k0+8..k0+15, k0+24..k0+31,
    // matching the A layout, from the row-major f16 codebook.
    const _Float16* brow = cb16 + ((size_t)l15) * C_ + hi * 8;

    for (int ct = 0; ct < KCB / 16; ++ct) {
        int code = ct * 16 + l15;
        float cn = cbnorm[code];
        v8f acc0 = {};
        v8f acc1 = {};
#pragma unroll
        for (int kk = 0; kk < 8; ++kk) {
            v8h blo = *(const v8h*)(brow + kk * 32);
            v8h bhi = *(const v8h*)(brow + kk * 32 + 16);
            v16h Bt;
#pragma unroll
            for (int j = 0; j < 8; ++j) { Bt[j] = blo[j]; Bt[j + 8] = bhi[j]; }
            acc0 = __builtin_amdgcn_wmma_f32_16x16x32_f16(
                false, A0[kk], false, Bt, (short)0, acc0, false, false);
            acc1 = __builtin_amdgcn_wmma_f32_16x16x32_f16(
                false, A1[kk], false, Bt, (short)0, acc1, false, false);
        }
        // ---- issue-order pipeline: loads run 2 fragments ahead of WMMAs ----
        // [cbnorm + frag0 + frag1 loads]
        __builtin_amdgcn_sched_group_barrier(SGB_VMEM_READ, 5, 0);
#pragma unroll
        for (int g = 0; g < 6; ++g) {
            __builtin_amdgcn_sched_group_barrier(SGB_WMMA, 2, 0);       // pair g
            __builtin_amdgcn_sched_group_barrier(SGB_VMEM_READ, 2, 0);  // frag g+2
        }
        __builtin_amdgcn_sched_group_barrier(SGB_WMMA, 2, 0);           // pair 6
        __builtin_amdgcn_sched_group_barrier(SGB_WMMA, 2, 0);           // pair 7

        brow += 16 * C_;
        // d' = ||c||^2 - 2 x.c   (||x||^2 constant per row -> dropped for argmin)
#pragma unroll
        for (int r = 0; r < 8; ++r) {
            float v0 = fmaf(-2.0f, acc0[r], cn);
            if (v0 < best[r])     { best[r]     = v0; bidx[r]     = code; }
            float v1 = fmaf(-2.0f, acc1[r], cn);
            if (v1 < best[8 + r]) { best[8 + r] = v1; bidx[8 + r] = code; }
        }
    }

    // ---- cross-lane (16-wide) argmin reduction; D layout: VGPR r holds
    // pixel M = r (lanes 0-15) / M = 8+r (lanes 16-31), N = l15.
#pragma unroll
    for (int g = 0; g < 16; ++g) {
        float v = best[g];
        int   id = bidx[g];
#pragma unroll
        for (int m = 1; m < 16; m <<= 1) {
            float ov = __shfl_xor(v, m, 16);
            int   oi = __shfl_xor(id, m, 16);
            if (ov < v || (ov == v && oi < id)) { v = ov; id = oi; }
        }
        if (l15 == 0) {
            int r    = g & 7;
            int half = g >> 3;              // 0: pixels 0..15, 1: pixels 16..31
            int pix  = tile * 32 + half * 16 + r + hi * 8;
            idx_ws[pix]  = id;
            out_idx[pix] = (float)id;
        }
    }
}

// ---------------------------------------------------------------------------
// 4) gather z_q, write u-copy + z_train, accumulate sum((z_q - u)^2)
// ---------------------------------------------------------------------------
__global__ void vq_finalize(const float* __restrict__ u,
                            const float* __restrict__ cb,
                            const int* __restrict__ idx_ws,
                            float* __restrict__ out_u,
                            float* __restrict__ out_zt,
                            float* __restrict__ loss_acc) {
    __shared__ float red[256];
    size_t e = (size_t)blockIdx.x * 256 + threadIdx.x;
    int b  = (int)(e >> 18);          // C_*HW_ = 2^18
    int c  = (int)((e >> 10) & 255);
    int hw = (int)(e & 1023);
    int n  = (b << 10) | hw;
    int id = idx_ws[n];
    float zq = cb[(size_t)id * C_ + c];
    float uv = u[e];
    out_u[e]  = uv;
    out_zt[e] = zq;     // z_train == z_q numerically (u - stop_grad(u) == 0)
    float d = zq - uv;
    red[threadIdx.x] = d * d;
    __syncthreads();
    for (int s = 128; s > 0; s >>= 1) {
        if (threadIdx.x < s) red[threadIdx.x] += red[threadIdx.x + s];
        __syncthreads();
    }
    if (threadIdx.x == 0) atomicAdd(loss_acc, red[0]);
}

__global__ void vq_write_loss(const float* __restrict__ loss_acc,
                              float* __restrict__ out_loss) {
    // codebook_loss + BETA*commitment_loss = (1 + 0.25) * mean((z_q - u)^2)
    *out_loss = 1.25f * (*loss_acc) / (float)NELEM;
}

// ---------------------------------------------------------------------------
extern "C" void kernel_launch(void* const* d_in, const int* in_sizes, int n_in,
                              void* d_out, int out_size, void* d_ws, size_t ws_size,
                              hipStream_t stream) {
    const float* u  = (const float*)d_in[0];
    const float* cb = (const float*)d_in[1];
    float* out = (float*)d_out;

    char* ws = (char*)d_ws;
    _Float16* u16    = (_Float16*)(ws + WS_U16);
    _Float16* cb16   = (_Float16*)(ws + WS_CB16);
    float*    cbnorm = (float*)   (ws + WS_CBNORM);
    int*      idx_ws = (int*)     (ws + WS_IDX);
    float*    lossa  = (float*)   (ws + WS_LOSS);

    vq_prep_cb<<<KCB, 256, 0, stream>>>(cb, cb16, cbnorm, lossa);

    dim3 tgrid(HW_ / 32, C_ / 32, B_);
    vq_transpose_u<<<tgrid, dim3(32, 8), 0, stream>>>(u, u16);

    vq_main<<<NPIX / (32 * 8), 256, 0, stream>>>(u16, cb16, cbnorm,
                                                 idx_ws, out + OUT_IDX);

    vq_finalize<<<NELEM / 256, 256, 0, stream>>>(u, cb, idx_ws,
                                                 out + OUT_U, out + OUT_ZT, lossa);

    vq_write_loss<<<1, 1, 0, stream>>>(lossa, out + OUT_LOSS);
}
